// AutoRegressiveStochasticActor_24240795419013
// MI455X (gfx1250) — compile-verified, matrix operand order checked by reading, not run on hardware
//
#include <hip/hip_runtime.h>
#include <hip/hip_bf16.h>
#include <math.h>

// ---------------------------------------------------------------------------
// CDNA5 / gfx1250 fused autoregressive GRU actor, bf16 WMMA (16x16x32) tiles.
// wave32; block = 256 threads = 8 waves; each wave owns MT=4 16-row M-tiles.
// GRU-step kernel stages its 6 weight panels in LDS (DScnt) so activation
// streams (LOADcnt) pipeline independently.
// ---------------------------------------------------------------------------

typedef __bf16 bf16_t;
typedef __attribute__((ext_vector_type(8)))  bf16_t v8bf;
typedef __attribute__((ext_vector_type(16))) bf16_t v16bf;
typedef __attribute__((ext_vector_type(8)))  float  v8f;

#define B_ROWS 16384
#define D_IN   376
#define N_ACT  17
#define HID    400
#define NBAS   64
#define K_SE   384   // D padded to multiple of 32
#define K_H    416   // H padded to multiple of 32
#define LDH    416
#define MT     4     // M-tiles per wave (64 rows/wave, 512 rows/block)
#define LDS_LDK (K_H + 8)                 // 424: pad to stagger LDS banks
#define GRU_SMEM_BYTES (6 * 16 * LDS_LDK * (int)sizeof(bf16_t))   // 81,408 B
#define PI_F   3.14159265358979323846f

__device__ __forceinline__ v8f wmma_bf16(v16bf a, v16bf b, v8f c) {
  // v_wmma_f32_16x16x32_bf16  D = A(16x32) * B(32x16) + C(16x16 f32)
  return __builtin_amdgcn_wmma_f32_16x16x32_bf16(false, a, false, b, (short)0, c,
                                                 false, false);
}

// ISA fragment layout (16-bit A 16x32 / B 32x16): lane L holds row/col (L&15),
// K-chunks of 8 at k0 + (L>>4)*8 and k0 + (L>>4)*8 + 16. Both A (row-major
// activations [M][K]) and B (torch weight [N][K]) load identically.
__device__ __forceinline__ v16bf frag_ld(const bf16_t* __restrict__ p, int ld,
                                         int rbase, int k0, int lane) {
  int r  = rbase + (lane & 15);
  int kc = k0 + ((lane >> 4) << 3);
  const bf16_t* q = p + (size_t)r * ld + kc;
  v8bf lo = *(const v8bf*)q;
  v8bf hi = *(const v8bf*)(q + 16);
  v16bf out;
#pragma unroll
  for (int i = 0; i < 8; ++i) { out[i] = lo[i]; out[i + 8] = hi[i]; }
  return out;
}

// Same fragment gather, but from an LDS-resident panel array (emits ds_load).
__device__ __forceinline__ v16bf frag_lds(const bf16_t* s, int panel, int k0, int lane) {
  int r  = lane & 15;
  int kc = k0 + ((lane >> 4) << 3);
  const bf16_t* q = s + (panel * 16 + r) * LDS_LDK + kc;
  v8bf lo = *(const v8bf*)q;
  v8bf hi = *(const v8bf*)(q + 16);
  v16bf out;
#pragma unroll
  for (int i = 0; i < 8; ++i) { out[i] = lo[i]; out[i + 8] = hi[i]; }
  return out;
}

__device__ __forceinline__ float lrelu_f(float v) { return v >= 0.f ? v : 0.01f * v; }
__device__ __forceinline__ float sigmoid_f(float x) { return 1.f / (1.f + __expf(-x)); }
__device__ __forceinline__ float tanh_f(float x) {
  float ax = fabsf(x);
  float e  = __expf(-2.f * ax);
  float t  = (1.f - e) / (1.f + e);
  return copysignf(t, x);
}

// -------------------------- utility kernels --------------------------------

__global__ void zero_kernel(uint4* __restrict__ p, size_t n16) {
  size_t i = (size_t)blockIdx.x * blockDim.x + threadIdx.x;
  size_t stride = (size_t)gridDim.x * blockDim.x;
  uint4 z; z.x = z.y = z.z = z.w = 0u;
  for (; i < n16; i += stride) p[i] = z;
}

// dst[n][k] = (k<Kin ? bf16(src[n][srcOff+k]) : 0), dst is [N][Kpad] row-major.
__global__ void pack_bf16_kernel(bf16_t* __restrict__ dst, const float* __restrict__ src,
                                 int N, int srcLd, int srcOff, int Kin, int Kpad) {
  size_t total = (size_t)N * Kpad;
  size_t i = (size_t)blockIdx.x * blockDim.x + threadIdx.x;
  size_t stride = (size_t)gridDim.x * blockDim.x;
  for (; i < total; i += stride) {
    int n = (int)(i / Kpad);
    int k = (int)(i % Kpad);
    float v = (k < Kin) ? src[(size_t)n * srcLd + srcOff + k] : 0.f;
    dst[i] = (bf16_t)v;
  }
}

// -------------------------- GEMM kernels -----------------------------------

// C[row, ncol0+0..15] = act( A[row,:] @ B[col,:] + bias[col] ); bf16 and/or f32
// out. B-fragment loaded once per k-step, reused across MT M-tiles.
__global__ void gemm_tile_kernel(const bf16_t* __restrict__ A, const bf16_t* __restrict__ Bm,
                                 const float* __restrict__ bias, int ldK, int nK,
                                 bf16_t* __restrict__ outBf, int ldOutBf,
                                 float* __restrict__ outF, int ldOutF, int act) {
  int lane = threadIdx.x & 31, wave = threadIdx.x >> 5;
  int rbase = blockIdx.x * (128 * MT) + wave * (16 * MT);
  int ncol0 = blockIdx.y * 16;
  v8f acc[MT];
  v8f z = {};
#pragma unroll
  for (int mt = 0; mt < MT; ++mt) acc[mt] = z;
  for (int ks = 0; ks < nK; ++ks) {
    int k0 = ks * 32;
    v16bf b = frag_ld(Bm, ldK, ncol0, k0, lane);
    v16bf a[MT];
#pragma unroll
    for (int mt = 0; mt < MT; ++mt) a[mt] = frag_ld(A, ldK, rbase + 16 * mt, k0, lane);
#pragma unroll
    for (int mt = 0; mt < MT; ++mt) acc[mt] = wmma_bf16(a[mt], b, acc[mt]);
  }
  int col = ncol0 + (lane & 15);
  float bv = bias ? bias[col] : 0.f;
#pragma unroll
  for (int mt = 0; mt < MT; ++mt) {
    int rr = rbase + 16 * mt + ((lane >> 4) << 3);
#pragma unroll
    for (int e = 0; e < 8; ++e) {
      float v = acc[mt][e] + bv;
      if (act) v = lrelu_f(v);
      size_t row = (size_t)(rr + e);
      if (outBf) outBf[row * ldOutBf + col] = (bf16_t)v;
      if (outF)  outF[row * ldOutF + col]  = v;
    }
  }
}

// cosine-basis embedding + linear:
//   out[row,col] = act( sum_k cos(x[row]*pi*(k+1)) * W[col,k] + bias[col] )
// A-fragment (cos values) generated in-register in the exact WMMA lane layout.
__global__ void coslin_kernel(const float* __restrict__ x, int xStride,
                              const bf16_t* __restrict__ W, const float* __restrict__ bias,
                              bf16_t* __restrict__ out, int applyLrelu) {
  int lane = threadIdx.x & 31, wave = threadIdx.x >> 5;
  int rbase = blockIdx.x * (128 * MT) + wave * (16 * MT);
  int ncol0 = blockIdx.y * 16;
  float xp[MT];
#pragma unroll
  for (int mt = 0; mt < MT; ++mt)
    xp[mt] = x[(size_t)(rbase + 16 * mt + (lane & 15)) * xStride] * PI_F;
  v8f acc[MT];
  v8f zz = {};
#pragma unroll
  for (int mt = 0; mt < MT; ++mt) acc[mt] = zz;
#pragma unroll
  for (int ks = 0; ks < 2; ++ks) {           // K = 64 exactly
    int kc = ks * 32 + ((lane >> 4) << 3);
    v16bf b = frag_ld(W, NBAS, ncol0, ks * 32, lane);
#pragma unroll
    for (int mt = 0; mt < MT; ++mt) {
      v16bf a;
#pragma unroll
      for (int i = 0; i < 8; ++i) {
        a[i]     = (bf16_t)__cosf(xp[mt] * (float)(kc + i + 1));
        a[i + 8] = (bf16_t)__cosf(xp[mt] * (float)(kc + 16 + i + 1));
      }
      acc[mt] = wmma_bf16(a, b, acc[mt]);
    }
  }
  int col = ncol0 + (lane & 15);
  float bv = bias[col];
#pragma unroll
  for (int mt = 0; mt < MT; ++mt) {
    int rr = rbase + 16 * mt + ((lane >> 4) << 3);
#pragma unroll
    for (int e = 0; e < 8; ++e) {
      float v = acc[mt][e] + bv;
      if (applyLrelu) v = lrelu_f(v);
      out[(size_t)(rr + e) * LDH + col] = (bf16_t)v;
    }
  }
}

// Fused GRU step for one 16-col tile. The six weight panels (r/z/n for W_hh and
// W_ih[:,H:]) for this col-tile are staged in LDS once per block; the k-loop
// then reads weights via DScnt (ds_load) while streaming h/ae via LOADcnt,
// issuing 24 WMMAs per k-step. Epilogue: gates + h_new + had = h_new * ne_t.
__global__ void gru_step_kernel(const float* __restrict__ gise, const bf16_t* __restrict__ ae,
                                const bf16_t* __restrict__ h_in,
                                const bf16_t* __restrict__ Wih_ae, const bf16_t* __restrict__ Whh,
                                const float* __restrict__ b_hh, const bf16_t* __restrict__ ne,
                                bf16_t* __restrict__ h_out, bf16_t* __restrict__ had) {
  extern __shared__ __align__(16) bf16_t smem[];   // 6 panels x 16 rows x LDS_LDK
  int lane = threadIdx.x & 31, wave = threadIdx.x >> 5;
  int rbase = blockIdx.x * (128 * MT) + wave * (16 * MT);
  int ncol0 = blockIdx.y * 16;

  // Cooperative panel fill: panels 0..2 = Whh rows {0,H,2H}+ncol0, 3..5 = Wih_ae.
  {
    const int CPR = K_H / 8;                 // 52 v8bf chunks per row
    const int TOT = 6 * 16 * CPR;            // 4992 chunks
    for (int idx = threadIdx.x; idx < TOT; idx += 256) {
      int c   = idx % CPR;
      int rem = idx / CPR;                   // 0..95
      int r   = rem & 15;
      int p   = rem >> 4;                    // 0..5
      int pp  = (p < 3) ? p : p - 3;
      const bf16_t* src = ((p < 3) ? Whh : Wih_ae) +
                          (size_t)(pp * HID + ncol0 + r) * K_H + c * 8;
      *(v8bf*)(smem + (p * 16 + r) * LDS_LDK + c * 8) = *(const v8bf*)src;
    }
  }
  __syncthreads();

  v8f acc_r[MT], acc_z[MT], acc_gin[MT], acc_ghn[MT];
  v8f zz = {};
#pragma unroll
  for (int mt = 0; mt < MT; ++mt) {
    acc_r[mt] = zz; acc_z[mt] = zz; acc_gin[mt] = zz; acc_ghn[mt] = zz;
  }
  for (int ks = 0; ks < 13; ++ks) {          // K = 416 (H padded)
    int k0 = ks * 32;
    // batch the k-step's global loads (one clause; single wait for 24 WMMAs)
    v16bf a_h[MT], a_ae[MT];
#pragma unroll
    for (int mt = 0; mt < MT; ++mt) {
      int rb = rbase + 16 * mt;
      a_h[mt]  = frag_ld(h_in, LDH, rb, k0, lane);
      a_ae[mt] = frag_ld(ae,   LDH, rb, k0, lane);
    }
    v16bf b_hr = frag_lds(smem, 0, k0, lane);
    v16bf b_hz = frag_lds(smem, 1, k0, lane);
    v16bf b_hn = frag_lds(smem, 2, k0, lane);
    v16bf b_ir = frag_lds(smem, 3, k0, lane);
    v16bf b_iz = frag_lds(smem, 4, k0, lane);
    v16bf b_in = frag_lds(smem, 5, k0, lane);
#pragma unroll
    for (int mt = 0; mt < MT; ++mt) {
      acc_r[mt]   = wmma_bf16(a_h[mt],  b_hr, acc_r[mt]);
      acc_r[mt]   = wmma_bf16(a_ae[mt], b_ir, acc_r[mt]);
      acc_z[mt]   = wmma_bf16(a_h[mt],  b_hz, acc_z[mt]);
      acc_z[mt]   = wmma_bf16(a_ae[mt], b_iz, acc_z[mt]);
      acc_ghn[mt] = wmma_bf16(a_h[mt],  b_hn, acc_ghn[mt]);
      acc_gin[mt] = wmma_bf16(a_ae[mt], b_in, acc_gin[mt]);
    }
  }
  int col = ncol0 + (lane & 15);
  float bhr = b_hh[col], bhz = b_hh[HID + col], bhn = b_hh[2 * HID + col];
#pragma unroll
  for (int mt = 0; mt < MT; ++mt) {
    int rr = rbase + 16 * mt + ((lane >> 4) << 3);
#pragma unroll
    for (int e = 0; e < 8; ++e) {
      size_t row = (size_t)(rr + e);
      const float* g = gise + row * 1200;    // gi_se already contains b_ih
      float gr   = acc_r[mt][e]   + g[col]           + bhr;
      float gz   = acc_z[mt][e]   + g[HID + col]     + bhz;
      float gn_i = acc_gin[mt][e] + g[2 * HID + col];
      float gn_h = acc_ghn[mt][e] + bhn;
      float r = sigmoid_f(gr);
      float z = sigmoid_f(gz);
      float n = tanh_f(gn_i + r * gn_h);
      float hprev = (float)h_in[row * LDH + col];
      float hnew  = (1.f - z) * n + z * hprev;
      h_out[row * LDH + col] = (bf16_t)hnew;
      float hd = hnew * (float)ne[row * LDH + col];
      had[row * LDH + col] = (bf16_t)hd;
    }
  }
}

// Fused head: l1 = lrelu(had @ W_l1^T + b_l1); a = tanh(l1 . w_l2 + b_l2).
// Per-row dot folded into the N-tile loop; cross-lane shfl_xor reduction.
__global__ void head_kernel(const bf16_t* __restrict__ had, const bf16_t* __restrict__ Wl1,
                            const float* __restrict__ b_l1, const float* __restrict__ w_l2,
                            const float* __restrict__ b_l2, float* __restrict__ out,
                            float* __restrict__ act_buf, int t) {
  int lane = threadIdx.x & 31, wave = threadIdx.x >> 5;
  int rbase = blockIdx.x * (128 * MT) + wave * (16 * MT);
  float part[MT][8];
#pragma unroll
  for (int mt = 0; mt < MT; ++mt)
#pragma unroll
    for (int e = 0; e < 8; ++e) part[mt][e] = 0.f;
  for (int j = 0; j < 25; ++j) {
    int ncol0 = j * 16;
    v8f acc[MT];
    v8f zz = {};
#pragma unroll
    for (int mt = 0; mt < MT; ++mt) acc[mt] = zz;
    for (int ks = 0; ks < 13; ++ks) {
      int k0 = ks * 32;
      v16bf b = frag_ld(Wl1, K_H, ncol0, k0, lane);
      v16bf a[MT];
#pragma unroll
      for (int mt = 0; mt < MT; ++mt) a[mt] = frag_ld(had, LDH, rbase + 16 * mt, k0, lane);
#pragma unroll
      for (int mt = 0; mt < MT; ++mt) acc[mt] = wmma_bf16(a[mt], b, acc[mt]);
    }
    int col = ncol0 + (lane & 15);
    float bv = b_l1[col], wv = w_l2[col];
#pragma unroll
    for (int mt = 0; mt < MT; ++mt)
#pragma unroll
      for (int e = 0; e < 8; ++e) part[mt][e] += lrelu_f(acc[mt][e] + bv) * wv;
  }
#pragma unroll
  for (int mt = 0; mt < MT; ++mt)
#pragma unroll
    for (int e = 0; e < 8; ++e) {
      part[mt][e] += __shfl_xor(part[mt][e], 1, 32);
      part[mt][e] += __shfl_xor(part[mt][e], 2, 32);
      part[mt][e] += __shfl_xor(part[mt][e], 4, 32);
      part[mt][e] += __shfl_xor(part[mt][e], 8, 32);
    }
  if ((lane & 15) == 0) {
    float bl2 = b_l2[0];
#pragma unroll
    for (int mt = 0; mt < MT; ++mt) {
      int rr = rbase + 16 * mt + ((lane >> 4) << 3);
#pragma unroll
      for (int e = 0; e < 8; ++e) {
        float a = tanh_f(part[mt][e] + bl2);
        out[(size_t)(rr + e) * N_ACT + t] = a;
        act_buf[rr + e] = a;
      }
    }
  }
}

// ---------------------------------------------------------------------------

extern "C" void kernel_launch(void* const* d_in, const int* in_sizes, int n_in,
                              void* d_out, int out_size, void* d_ws, size_t ws_size,
                              hipStream_t stream) {
  (void)in_sizes; (void)n_in; (void)out_size; (void)ws_size;
  const float* state = (const float*)d_in[0];
  const float* taus  = (const float*)d_in[1];
  const float* W_se  = (const float*)d_in[2];
  const float* b_se  = (const float*)d_in[3];
  const float* W_ne  = (const float*)d_in[4];
  const float* b_ne  = (const float*)d_in[5];
  const float* W_ae  = (const float*)d_in[6];
  const float* b_ae  = (const float*)d_in[7];
  const float* W_ih  = (const float*)d_in[8];
  const float* W_hh  = (const float*)d_in[9];
  const float* b_ih  = (const float*)d_in[10];
  const float* b_hh  = (const float*)d_in[11];
  const float* W_l1  = (const float*)d_in[12];
  const float* b_l1  = (const float*)d_in[13];
  const float* W_l2  = (const float*)d_in[14];
  const float* b_l2  = (const float*)d_in[15];
  float* out = (float*)d_out;

  char* ws = (char*)d_ws;
  size_t off = 0;
  auto alloc = [&](size_t bytes) -> char* {
    char* p = ws + off;
    off = (off + bytes + 255) & ~(size_t)255;
    return p;
  };
  const size_t actBytes = (size_t)B_ROWS * LDH * sizeof(bf16_t);
  bf16_t* se_bf   = (bf16_t*)alloc(actBytes);
  bf16_t* ne_bf   = (bf16_t*)alloc(actBytes);
  bf16_t* ae_bf   = (bf16_t*)alloc(actBytes);
  bf16_t* had_bf  = (bf16_t*)alloc(actBytes);
  bf16_t* h0      = (bf16_t*)alloc(actBytes);
  bf16_t* h1      = (bf16_t*)alloc(actBytes);
  float*  act_buf = (float*)alloc((size_t)B_ROWS * sizeof(float));
  size_t zeroBytes = off;   // zero every launch: pads stay 0, h0 = 0, a0 = 0
  bf16_t* st_bf   = (bf16_t*)alloc((size_t)B_ROWS * K_SE * sizeof(bf16_t));
  float*  gi_se   = (float*)alloc((size_t)B_ROWS * 1200 * sizeof(float));
  bf16_t* Wse_p   = (bf16_t*)alloc((size_t)HID * K_SE * sizeof(bf16_t));
  bf16_t* Wne_p   = (bf16_t*)alloc((size_t)HID * NBAS * sizeof(bf16_t));
  bf16_t* Wae_p   = (bf16_t*)alloc((size_t)HID * NBAS * sizeof(bf16_t));
  bf16_t* Wihse_p = (bf16_t*)alloc((size_t)1200 * K_H * sizeof(bf16_t));
  bf16_t* Wihae_p = (bf16_t*)alloc((size_t)1200 * K_H * sizeof(bf16_t));
  bf16_t* Whh_p   = (bf16_t*)alloc((size_t)1200 * K_H * sizeof(bf16_t));
  bf16_t* Wl1_p   = (bf16_t*)alloc((size_t)HID * K_H * sizeof(bf16_t));

  dim3 blk(256);
  const int gx = B_ROWS / (128 * MT);       // 32 row-blocks

  // 0) zero h/pads/action (determinism under graph replay)
  zero_kernel<<<2048, blk, 0, stream>>>((uint4*)ws, zeroBytes / 16);

  // 1) pack fp32 -> bf16 with K padding (weights are [out][in] = ISA B layout)
  auto pack = [&](bf16_t* dst, const float* src, int N, int srcLd, int srcOff,
                  int Kin, int Kpad) {
    size_t total = (size_t)N * Kpad;
    int blocks = (int)((total + 256 * 8 - 1) / (256 * 8));
    if (blocks > 4096) blocks = 4096;
    if (blocks < 1) blocks = 1;
    pack_bf16_kernel<<<blocks, blk, 0, stream>>>(dst, src, N, srcLd, srcOff, Kin, Kpad);
  };
  pack(st_bf,   state, B_ROWS, D_IN, 0,   D_IN, K_SE);
  pack(Wse_p,   W_se,  HID,    D_IN, 0,   D_IN, K_SE);
  pack(Wne_p,   W_ne,  HID,    NBAS, 0,   NBAS, NBAS);
  pack(Wae_p,   W_ae,  HID,    NBAS, 0,   NBAS, NBAS);
  pack(Wihse_p, W_ih,  1200,   800,  0,   HID,  K_H);
  pack(Wihae_p, W_ih,  1200,   800,  HID, HID,  K_H);
  pack(Whh_p,   W_hh,  1200,   HID,  0,   HID,  K_H);
  pack(Wl1_p,   W_l1,  HID,    HID,  0,   HID,  K_H);

  // 2) se = lrelu(state @ W_se^T + b_se)           [B,400] bf16 (ld 416)
  gemm_tile_kernel<<<dim3(gx, HID / 16), blk, 0, stream>>>(
      st_bf, Wse_p, b_se, K_SE, K_SE / 32, se_bf, LDH, nullptr, 0, 1);

  // 3) gi_se = se @ W_ih[:, :H]^T + b_ih           [B,1200] f32 (loop-invariant)
  gemm_tile_kernel<<<dim3(gx, 1200 / 16), blk, 0, stream>>>(
      se_bf, Wihse_p, b_ih, K_H, K_H / 32, nullptr, 0, gi_se, 1200, 0);

  // 4) sequential scan over A=17 steps
  bf16_t* h_in = h0;
  bf16_t* h_out = h1;
  for (int t = 0; t < N_ACT; ++t) {
    // ne_t = cos_lin(taus[:, t])  (recomputed per step; avoids 446 MB buffer)
    coslin_kernel<<<dim3(gx, HID / 16), blk, 0, stream>>>(
        taus + t, N_ACT, Wne_p, b_ne, ne_bf, 0);
    // ae = lrelu(cos_lin(action))
    coslin_kernel<<<dim3(gx, HID / 16), blk, 0, stream>>>(
        act_buf, 1, Wae_p, b_ae, ae_bf, 1);
    // GRU gates + h_new + had (weights staged in 81 KB dynamic LDS)
    gru_step_kernel<<<dim3(gx, HID / 16), blk, GRU_SMEM_BYTES, stream>>>(
        gi_se, ae_bf, h_in, Wihae_p, Whh_p, b_hh, ne_bf, h_out, had_bf);
    // head: lrelu(W_l1) -> dot w_l2 -> tanh -> out[:, t] and next action
    head_kernel<<<dim3(gx, 1), blk, 0, stream>>>(
        had_bf, Wl1_p, b_l1, W_l2, b_l2, out, act_buf, t);
    bf16_t* tmp = h_in; h_in = h_out; h_out = tmp;
  }
}
